// AttentionHead_63410897158393
// MI455X (gfx1250) — matrix operand order repaired and checked by reading, not compiled, not run on hardware
//
#include <hip/hip_runtime.h>

// ---------------------------------------------------------------------------
// MI455X (gfx1250) attention head: bf16 WMMA (16x16x32) with fp32 accumulate.
// Memory roof dominates (134 MB probs write) -> write probs exactly once,
// read x exactly once. Features:
//   * async global->LDS staging (global_load_async_to_lds_b128 + ASYNCcnt)
//     of the weight fragments, double-buffered, shared by all 8 waves
//   * register double-buffering of every fragment stream (x, W, k, P, v)
//     so loads issue ahead of the consuming v_wmma (no s_wait_*cnt 0 stalls)
// ---------------------------------------------------------------------------

typedef __attribute__((ext_vector_type(16))) __bf16        v16bf;
typedef __attribute__((ext_vector_type(8)))  float         v8f;
typedef __attribute__((ext_vector_type(4)))  unsigned int  u32x4;
typedef __attribute__((ext_vector_type(2)))  float         f32x2;

#define B_ 8
#define S_ 2048
#define H_ 768
#define D_ 64
#define SROW (S_ + 16)   // padded LDS row (elems) to break bank conflicts

__device__ __forceinline__ unsigned int f2bf1(float f) {
  unsigned int x = __float_as_uint(f);
  return ((x + 0x7fffu + ((x >> 16) & 1u)) >> 16) & 0xffffu;  // RNE truncate
}
__device__ __forceinline__ unsigned int pack_bf2(float lo, float hi) {
  return f2bf1(lo) | (f2bf1(hi) << 16);
}

union Frag { unsigned int u[8]; u32x4 v[2]; v16bf h; };

__device__ __forceinline__ v8f wmma_bf16(const Frag& a, const Frag& b, v8f c) {
  return __builtin_amdgcn_wmma_f32_16x16x32_bf16(false, a.h, false, b.h,
                                                 (short)0, c, false, false);
}

// CDNA5 async global->LDS copy (16B per lane), tracked by ASYNCcnt.
__device__ __forceinline__ void async_g2lds_b128(unsigned int lds_byte_off,
                                                 const void* gaddr) {
  asm volatile("global_load_async_to_lds_b128 %0, %1, off"
               :: "v"(lds_byte_off), "v"(gaddr) : "memory");
}
__device__ __forceinline__ void wait_asynccnt0() {
  asm volatile("s_wait_asynccnt 0x0" ::: "memory");
}
// Flat shared pointer -> wave-relative LDS byte offset (low 32 bits of flat).
__device__ __forceinline__ unsigned int lds_off(const void* p) {
  return (unsigned int)(unsigned long long)p;
}

// ---------------------------------------------------------------------------
// Kernel 0: pre-swizzle Wq|Wk|Wv (768x64 fp32 row-major) into bf16 B-fragment
// layout ordered [kt][wi][nt] so each K-step slice is one contiguous 12 KB
// block for async staging. Per frag: 32 lanes x 8 dwords.
// Lane (c,h): column n = nt*16+c, K pairs = kt*32 + h*16 + 2j.
// ---------------------------------------------------------------------------
__global__ void qkv_wswizzle(const float* __restrict__ Wq, const float* __restrict__ Wk,
                             const float* __restrict__ Wv, unsigned int* __restrict__ wfrag) {
  int f    = blockIdx.x * 8 + (threadIdx.x >> 5);   // 288 fragments, linear in [kt][wi][nt]
  int lane = threadIdx.x & 31;
  int c = lane & 15, h = lane >> 4;
  int kt = f / 12, rem = f % 12, wi = rem >> 2, nt = rem & 3;
  const float* W = (wi == 0) ? Wq : (wi == 1) ? Wk : Wv;
  int n = nt * 16 + c;
  unsigned int* dst = wfrag + f * 256 + lane * 8;
#pragma unroll
  for (int j = 0; j < 8; ++j) {
    int k = kt * 32 + h * 16 + 2 * j;
    dst[j] = pack_bf2(W[k * 64 + n], W[(k + 1) * 64 + n]);
  }
}

// ---------------------------------------------------------------------------
// Kernel 1: fused QKV projection. One wave per 16-row tile of x ([16384,768]).
// Weight fragments async-staged to LDS (double-buffered across K-steps,
// shared by all 8 waves); x A-fragments prefetched one K-step ahead; the 12
// B-fragment ds_load streams are 2-deep pipelined against the WMMAs.
// Outputs:
//   qbf, kbf : [B*S, 64] bf16 row-major
//   vtbf     : [B, 64, S] bf16 (v transposed -> contiguous PV B-frags)
// ---------------------------------------------------------------------------
__global__ __launch_bounds__(256, 1)
void qkv_proj(const float* __restrict__ x, const unsigned int* __restrict__ wfrag,
              const float* __restrict__ bq, const float* __restrict__ bk,
              const float* __restrict__ bv,
              unsigned short* __restrict__ qbf, unsigned short* __restrict__ kbf,
              unsigned short* __restrict__ vtbf) {
  __shared__ unsigned int sW[2][12 * 256];          // 2 x 12KB fragment slices

  int tid  = threadIdx.x;
  int wvid = tid >> 5, lane = tid & 31;
  int mt = blockIdx.x * 8 + wvid;                   // 0..1023 M-tiles
  int c = lane & 15, h = lane >> 4;
  const float* xrow = x + (size_t)(mt * 16 + c) * H_;

  // Async-stage the [kt] weight-fragment slice into sW[buf] (whole block).
  auto stageW = [&](int kt, int buf) {
#pragma unroll
    for (int r = 0; r < 3; ++r) {
      int elt = r * 256 + tid;                      // 768 x 16B = 12 KB
      async_g2lds_b128(lds_off(&sW[buf][elt * 4]),
                       (const void*)(wfrag + (size_t)kt * 3072 + elt * 4));
    }
  };

  f32x2 xraw[2][8];
  auto loadX = [&](int kt, int buf) {
#pragma unroll
    for (int j = 0; j < 8; ++j) {
      int k0 = kt * 32 + 2 * (j + h * 8);  // contiguous K order (k = 16h + s), matches B
      xraw[buf][j] = *(const f32x2*)(xrow + k0);
    }
  };

  v8f acc[3][4] = {};

  stageW(0, 0);
  loadX(0, 0);
  wait_asynccnt0();
  __syncthreads();

  for (int kt = 0; kt < 24; ++kt) {
    int buf = kt & 1;
    if (kt < 23) {                                   // prefetch next K-step
      stageW(kt + 1, buf ^ 1);
      loadX(kt + 1, buf ^ 1);
    }
    Frag a;
#pragma unroll
    for (int j = 0; j < 8; ++j)
      a.u[j] = pack_bf2(xraw[buf][j].x, xraw[buf][j].y);

    // 12 fragments (f = wi*4+nt), 2-deep pipelined ds_load -> wmma.
    Frag bfr[2];
    auto loadB = [&](int f, int rb) {
      const unsigned int* src = &sW[buf][f * 256 + lane * 8];
      bfr[rb].v[0] = *(const u32x4*)(src);           // ds_load_b128
      bfr[rb].v[1] = *(const u32x4*)(src + 4);
    };
    loadB(0, 0);
#pragma unroll
    for (int f = 0; f < 12; ++f) {
      if (f < 11) loadB(f + 1, (f + 1) & 1);         // issue next pair first
      acc[f >> 2][f & 3] = wmma_bf16(a, bfr[f & 1], acc[f >> 2][f & 3]);
    }

    wait_asynccnt0();                                // next slice landed
    __syncthreads();                                 // everyone done with buf
  }

  int s0 = mt * 16;
#pragma unroll
  for (int wi = 0; wi < 3; ++wi) {
    const float* bias = (wi == 0) ? bq : (wi == 1) ? bk : bv;
#pragma unroll
    for (int nt = 0; nt < 4; ++nt) {
      int n = nt * 16 + c;
      float bval = bias[n];
#pragma unroll
      for (int i = 0; i < 8; ++i) {
        // D layout: VGPR i -> (M = i + h*8, N = c)
        float val = acc[wi][nt][i] + bval;
        int m = s0 + i + h * 8;
        unsigned short v16 = (unsigned short)f2bf1(val);
        if (wi == 0)      qbf[m * 64 + n] = v16;
        else if (wi == 1) kbf[m * 64 + n] = v16;
        else {
          int bb = m >> 11, sl = m & (S_ - 1);
          vtbf[((size_t)bb * 64 + n) * S_ + sl] = v16;
        }
      }
    }
  }
}

// ---------------------------------------------------------------------------
// Kernel 2: one workgroup per (batch, 16 query rows). 8 waves.
// Phase 1: scores = q @ k^T * 0.125 via WMMA (K frags double-buffered).
// Phase 2: fp32 softmax (LDS reductions + v_exp_f32), single probs write.
// Phase 3: out = P @ v via WMMA, split-K across waves, P/v double-buffered.
// ---------------------------------------------------------------------------
__global__ __launch_bounds__(256, 1)
void attn(const unsigned short* __restrict__ qbf_us, const unsigned short* __restrict__ kbf_us,
          const unsigned short* __restrict__ vtbf_us,
          float* __restrict__ probs, float* __restrict__ out) {
  extern __shared__ char smem[];
  float*          sS = (float*)smem;                                   // 16*SROW f32
  unsigned short* sP = (unsigned short*)(smem + 16 * SROW * 4);        // 16*SROW bf16
  float*          sR = (float*)(smem + 16 * SROW * 4 + 16 * SROW * 2); // 256 f32

  const unsigned int* qbf  = (const unsigned int*)qbf_us;
  const unsigned int* kbf  = (const unsigned int*)kbf_us;
  const unsigned int* vtbf = (const unsigned int*)vtbf_us;

  int wvid = threadIdx.x >> 5, lane = threadIdx.x & 31;
  int c = lane & 15, h = lane >> 4;
  int b = blockIdx.x >> 7, mt = blockIdx.x & 127;
  int s0 = mt * 16;

  // -------- phase 1: scores --------
  Frag aq[2];
  {
    int rowdw = (b * S_ + s0 + c) * 32;              // q row = 32 dwords bf16
#pragma unroll
    for (int ds = 0; ds < 2; ++ds) {
      aq[ds].v[0] = *(const u32x4*)(qbf + rowdw + ds * 16 + h * 8);
      aq[ds].v[1] = *(const u32x4*)(qbf + rowdw + ds * 16 + h * 8 + 4);
    }
  }
  {
    Frag bkf[2][2];                                  // [buf][ds]
    auto loadK = [&](int nti, int buf) {
      int t0 = (wvid * 16 + nti) * 16;
      int colrow = (b * S_ + t0 + c) * 32;           // B column = key t0+c
#pragma unroll
      for (int ds = 0; ds < 2; ++ds) {
        bkf[buf][ds].v[0] = *(const u32x4*)(kbf + colrow + ds * 16 + h * 8);
        bkf[buf][ds].v[1] = *(const u32x4*)(kbf + colrow + ds * 16 + h * 8 + 4);
      }
    };
    loadK(0, 0);
    for (int nti = 0; nti < 16; ++nti) {             // 16 key-tiles per wave
      int buf = nti & 1;
      if (nti < 15) loadK(nti + 1, buf ^ 1);         // prefetch next tile
      v8f acc = {};
      acc = wmma_bf16(aq[0], bkf[buf][0], acc);
      acc = wmma_bf16(aq[1], bkf[buf][1], acc);
      int t0 = (wvid * 16 + nti) * 16;
#pragma unroll
      for (int i = 0; i < 8; ++i)
        sS[(i + h * 8) * SROW + t0 + c] = acc[i] * 0.125f;   // 1/sqrt(64)
    }
  }
  __syncthreads();

  // -------- phase 2: softmax + probs write --------
  {
    int row = threadIdx.x >> 4, c16 = threadIdx.x & 15;   // 16 threads per row
    float* srow = sS + row * SROW;
    float mx = -3.402823e38f;
    for (int i = 0; i < 128; ++i) mx = fmaxf(mx, srow[c16 + i * 16]);
    sR[row * 16 + c16] = mx;
    __syncthreads();
    float rmax = -3.402823e38f;
#pragma unroll
    for (int i = 0; i < 16; ++i) rmax = fmaxf(rmax, sR[row * 16 + i]);
    __syncthreads();
    float sum = 0.f;
    for (int i = 0; i < 128; ++i) {
      int idx = c16 + i * 16;
      float e = __expf(srow[idx] - rmax);
      srow[idx] = e;
      sum += e;
    }
    sR[row * 16 + c16] = sum;
    __syncthreads();
    float tot = 0.f;
#pragma unroll
    for (int i = 0; i < 16; ++i) tot += sR[row * 16 + i];
    float inv = 1.f / tot;
    float* prow = probs + ((size_t)(b * S_ + s0 + row)) * S_;
    unsigned short* pbrow = sP + row * SROW;
    for (int i = 0; i < 128; ++i) {
      int idx = c16 + i * 16;
      float p = srow[idx] * inv;
      prow[idx] = p;                                 // dominant HBM write, once
      pbrow[idx] = (unsigned short)f2bf1(p);
    }
  }
  __syncthreads();

  // -------- phase 3: out = P @ v (split-K: wave = (ntile, K-half)) --------
  {
    int nt = wvid & 3, kh = wvid >> 2;
    const unsigned int* sPu = (const unsigned int*)sP;
    int d = nt * 16 + c;
    int vrow = (b * 64 + d) * (S_ / 2);              // v^T row in dwords
    v8f acc = {};
    Frag apf[2], bvf[2];
    auto loadPV = [&](int ks, int buf) {
      int prowdw = c * (SROW / 2) + ks * 16;         // P row c, K window ks
      apf[buf].v[0] = *(const u32x4*)(sPu + prowdw + h * 8);       // ds_load_b128
      apf[buf].v[1] = *(const u32x4*)(sPu + prowdw + h * 8 + 4);
      bvf[buf].v[0] = *(const u32x4*)(vtbf + vrow + ks * 16 + h * 8);
      bvf[buf].v[1] = *(const u32x4*)(vtbf + vrow + ks * 16 + h * 8 + 4);
    };
    int ks0 = kh * 32;
    loadPV(ks0, 0);
    for (int i = 0; i < 32; ++i) {
      int buf = i & 1;
      if (i < 31) loadPV(ks0 + i + 1, buf ^ 1);      // prefetch next K-step
      acc = wmma_bf16(apf[buf], bvf[buf], acc);
    }
    float* spart = sS;                               // reuse score strip
    if (kh == 1) {
#pragma unroll
      for (int i = 0; i < 8; ++i) spart[(nt * 8 + i) * 32 + lane] = acc[i];
    }
    __syncthreads();
    if (kh == 0) {
#pragma unroll
      for (int i = 0; i < 8; ++i) {
        float val = acc[i] + spart[(nt * 8 + i) * 32 + lane];
        out[((size_t)(b * S_ + s0 + i + h * 8)) * 64 + d] = val;
      }
    }
  }
}

// ---------------------------------------------------------------------------
extern "C" void kernel_launch(void* const* d_in, const int* in_sizes, int n_in,
                              void* d_out, int out_size, void* d_ws, size_t ws_size,
                              hipStream_t stream) {
  const float* x  = (const float*)d_in[0];
  const float* Wq = (const float*)d_in[1];
  const float* bq = (const float*)d_in[2];
  const float* Wk = (const float*)d_in[3];
  const float* bk = (const float*)d_in[4];
  const float* Wv = (const float*)d_in[5];
  const float* bv = (const float*)d_in[6];

  float* out   = (float*)d_out;
  float* probs = out + (size_t)B_ * S_ * D_;          // (out, probs) concatenated

  char* ws = (char*)d_ws;
  unsigned int*   wfrag = (unsigned int*)ws;                                // 288 KB
  unsigned short* qbf   = (unsigned short*)(ws + (512u << 10));             // 2 MB
  unsigned short* kbf   = (unsigned short*)(ws + (512u << 10) + (2u << 20));// 2 MB
  unsigned short* vtbf  = (unsigned short*)(ws + (512u << 10) + (4u << 20));// 2 MB

  qkv_wswizzle<<<36, 256, 0, stream>>>(Wq, Wk, Wv, wfrag);
  qkv_proj<<<128, 256, 0, stream>>>(x, wfrag, bq, bk, bv, qbf, kbf, vtbf);

  size_t smem2 = (size_t)16 * SROW * 4 + (size_t)16 * SROW * 2 + 1024;
  attn<<<dim3(B_ * (S_ / 16)), 256, (unsigned int)smem2, stream>>>(qbf, kbf, vtbf, probs, out);
}
